// VRNN_11278584119713
// MI455X (gfx1250) — compile-verified, matrix operand order checked by reading, not run on hardware
//
#include <hip/hip_runtime.h>
#include <cstdint>
#include <cstddef>

// ---------------------------------------------------------------------------
// Tree-GRU VAE for MI455X (gfx1250, wave32, WMMA bf16 16x16x32, f32 accum)
//
//   - gi = emb[nodes] @ Wih^T + bih precomputed for all 4 cells as big WMMA
//     GEMMs (parallel, off the critical path)
//   - per-step kernel fuses h @ Whh^T (6 gate tiles) + GRU nonlinearity +
//     edge select in registers (ping-pong h buffers for cross-block safety)
//   - logits GEMM 6144x6016(pad)x512 in bf16 WMMA (64x32 wave tiles to halve
//     A-operand L2 traffic), then log_softmax reading padded rows -> d_out
// ---------------------------------------------------------------------------

namespace cfg {
constexpr int S  = 48;
constexpr int D  = 128;
constexpr int H  = 512;
constexpr int U  = 512;
constexpr int L  = 256;
constexpr int V  = 6000;
constexpr int VP = 6016;          // V padded to a multiple of 32
constexpr int H3 = 3 * H;
constexpr int SD = S * D;
}

typedef __bf16 bf16_t;
typedef __attribute__((ext_vector_type(16))) __bf16 v16bf;
typedef __attribute__((ext_vector_type(8)))  float  v8f;
typedef __attribute__((ext_vector_type(4)))  float  f32x4;

// ---------------------------------------------------------------------------
// Fragment loader: 16x32 bf16 operand tile at (r0, k0) of a row-major K-major
// matrix (ld = ldk elements). Per CDNA5 ISA layout for 16-bit A/B operands:
//   lanes 0-15  : row = r0+lane,    halves 0-7 = K+0..7,  8-15 = K+16..23
//   lanes 16-31 : row = r0+lane-16, halves 0-7 = K+8..15, 8-15 = K+24..31
// Two 16-byte loads per lane. Requires ldk % 8 == 0 and 16B-aligned base.
// ---------------------------------------------------------------------------
__device__ __forceinline__ const bf16_t* frag_lane_ptr(const bf16_t* __restrict__ base,
                                                       int ldk) {
  const int lane = threadIdx.x & 31;
  return base + (size_t)(lane & 15) * (size_t)ldk + ((lane >> 4) << 3);
}

__device__ __forceinline__ v16bf load_frag(const bf16_t* __restrict__ base, int ldk) {
  const bf16_t* p = frag_lane_ptr(base, ldk);
  v16bf f;
  reinterpret_cast<f32x4*>(&f)[0] = *reinterpret_cast<const f32x4*>(p);
  reinterpret_cast<f32x4*>(&f)[1] = *reinterpret_cast<const f32x4*>(p + 16);
  return f;
}

__device__ __forceinline__ v8f wmma_bf16(v16bf a, v16bf b, v8f c) {
  return __builtin_amdgcn_wmma_f32_16x16x32_bf16(false, a, false, b, (short)0, c,
                                                 false, false);
}

__device__ __forceinline__ float sigmoidf_(float x) { return 1.0f / (1.0f + expf(-x)); }

// ---------------------------------------------------------------------------
// Generic C(M,N) = A(MxK) * B(NxK)^T + bias, A/B bf16, C fp32 (ldc stride).
// Block = 128 threads (4 waves). Wave computes 64(M) x 32(N) -> 8 WMMA per
// 6 fragment loads. Block tile = 256(M) x 32(N). Grid: (N/32, M/256).
// WGP-scope global_prefetch streams A/B two K-tiles ahead into near caches.
// ---------------------------------------------------------------------------
__global__ __launch_bounds__(128)
void gemm_nt_bf16(const bf16_t* __restrict__ A, const bf16_t* __restrict__ B,
                  const float* __restrict__ bias, float* __restrict__ C,
                  int M, int N, int K, int ldc) {
  (void)M; (void)N;
  const int wave = threadIdx.x >> 5;
  const int lane = threadIdx.x & 31;
  const int n0   = blockIdx.x * 32;
  const int m0   = blockIdx.y * 256 + wave * 64;
  const int col0 = n0 + (lane & 15);
  const int col1 = col0 + 16;

  const float bv0 = bias ? bias[col0] : 0.0f;
  const float bv1 = bias ? bias[col1] : 0.0f;
  v8f p0, p1, p2, p3, q0, q1, q2, q3;
#pragma unroll
  for (int i = 0; i < 8; ++i) {
    p0[i] = bv0; p1[i] = bv0; p2[i] = bv0; p3[i] = bv0;
    q0[i] = bv1; q1[i] = bv1; q2[i] = bv1; q3[i] = bv1;
  }

  const bf16_t* Ab = A + (size_t)m0 * K;
  const bf16_t* Bb = B + (size_t)n0 * K;
  for (int k0 = 0; k0 < K; k0 += 32) {
    v16bf b0 = load_frag(Bb + k0, K);
    v16bf b1 = load_frag(Bb + (size_t)16 * K + k0, K);
    v16bf a0 = load_frag(Ab + k0, K);
    v16bf a1 = load_frag(Ab + (size_t)16 * K + k0, K);
    v16bf a2 = load_frag(Ab + (size_t)32 * K + k0, K);
    v16bf a3 = load_frag(Ab + (size_t)48 * K + k0, K);
    // stream next-next K-tile into near caches (locality 3 -> WGP scope:
    // "pull into all cache levels on miss"); uniform branch keeps EXEC intact
    if (k0 + 64 < K) {
      __builtin_prefetch(frag_lane_ptr(Ab + k0 + 64, K), 0, 3);
      __builtin_prefetch(frag_lane_ptr(Bb + k0 + 64, K), 0, 3);
    }
    p0 = wmma_bf16(a0, b0, p0);
    p1 = wmma_bf16(a1, b0, p1);
    p2 = wmma_bf16(a2, b0, p2);
    p3 = wmma_bf16(a3, b0, p3);
    q0 = wmma_bf16(a0, b1, q0);
    q1 = wmma_bf16(a1, b1, q1);
    q2 = wmma_bf16(a2, b1, q2);
    q3 = wmma_bf16(a3, b1, q3);
  }

  // C/D layout: VGPR i -> (M = rsub + i, N = lane & 15), rsub = (lane>=16)*8
  const int rsub = (lane >> 4) << 3;
  float* Cp = C + (size_t)(m0 + rsub) * ldc + col0;
#pragma unroll
  for (int i = 0; i < 8; ++i) {
    Cp[(size_t)(i)      * ldc]      = p0[i];
    Cp[(size_t)(16 + i) * ldc]      = p1[i];
    Cp[(size_t)(32 + i) * ldc]      = p2[i];
    Cp[(size_t)(48 + i) * ldc]      = p3[i];
    Cp[(size_t)(i)      * ldc + 16] = q0[i];
    Cp[(size_t)(16 + i) * ldc + 16] = q1[i];
    Cp[(size_t)(32 + i) * ldc + 16] = q2[i];
    Cp[(size_t)(48 + i) * ldc + 16] = q3[i];
  }
}

// ---------------------------------------------------------------------------
// Fused GRU step: gh = hbf @ Whh^T for both branches (6 gate tiles), then
// r/z/n nonlinearity + edge select, writing h' (fp32 + bf16, ping-pong) and
// optionally the decoder output row (bf16 for the logits GEMM).
// Grid: H/16 blocks (j-tiles). Block: 256 threads = 8 waves; wave w owns
// rows d = 16w..16w+15 (all of D=128).
// ---------------------------------------------------------------------------
__global__ __launch_bounds__(256)
void rnn_step(int s,
              const float*  __restrict__ Gic, const float*  __restrict__ Gis,
              const bf16_t* __restrict__ Wc,  const bf16_t* __restrict__ Ws,
              const float*  __restrict__ bc,  const float*  __restrict__ bs,
              const unsigned char* __restrict__ edges,
              const float*  __restrict__ h_in,  const bf16_t* __restrict__ hbf_in,
              float*        __restrict__ h_out, bf16_t*       __restrict__ hbf_out,
              bf16_t*       __restrict__ outs) {
  using namespace cfg;
  const int wave = threadIdx.x >> 5;
  const int lane = threadIdx.x & 31;
  const int n0   = blockIdx.x * 16;     // j tile
  const int m0   = wave * 16;           // d tile
  const int col  = n0 + (lane & 15);    // j

  v8f acr, acz, acn, asr, asz, asn;
  {
    const float v0 = bc[col], v1 = bc[H + col], v2 = bc[2 * H + col];
    const float w0 = bs[col], w1 = bs[H + col], w2 = bs[2 * H + col];
#pragma unroll
    for (int i = 0; i < 8; ++i) {
      acr[i] = v0; acz[i] = v1; acn[i] = v2;
      asr[i] = w0; asz[i] = w1; asn[i] = w2;
    }
  }

  const bf16_t* Ab = hbf_in + (size_t)m0 * H;
  for (int k0 = 0; k0 < H; k0 += 32) {
    v16bf a   = load_frag(Ab + k0, H);
    v16bf wcr = load_frag(Wc + (size_t)(n0)         * H + k0, H);
    v16bf wcz = load_frag(Wc + (size_t)(H + n0)     * H + k0, H);
    v16bf wcn = load_frag(Wc + (size_t)(2 * H + n0) * H + k0, H);
    v16bf wsr = load_frag(Ws + (size_t)(n0)         * H + k0, H);
    v16bf wsz = load_frag(Ws + (size_t)(H + n0)     * H + k0, H);
    v16bf wsn = load_frag(Ws + (size_t)(2 * H + n0) * H + k0, H);
    acr = wmma_bf16(a, wcr, acr);
    acz = wmma_bf16(a, wcz, acz);
    acn = wmma_bf16(a, wcn, acn);
    asr = wmma_bf16(a, wsr, asr);
    asz = wmma_bf16(a, wsz, asz);
    asn = wmma_bf16(a, wsn, asn);
  }

  const int rsub  = (lane >> 4) << 3;
  const int gbase = s * D;
#pragma unroll
  for (int i = 0; i < 8; ++i) {
    const int d = m0 + rsub + i;
    const size_t g = (size_t)(gbase + d) * H3;
    const float hold = h_in[(size_t)d * H + col];
    // child branch
    float r  = sigmoidf_(Gic[g + col]         + acr[i]);
    float z  = sigmoidf_(Gic[g + H + col]     + acz[i]);
    float n  = tanhf(Gic[g + 2 * H + col] + r * acn[i]);
    float hc = (1.0f - z) * n + z * hold;
    // sibling branch
    float rs = sigmoidf_(Gis[g + col]         + asr[i]);
    float zs = sigmoidf_(Gis[g + H + col]     + asz[i]);
    float ns = tanhf(Gis[g + 2 * H + col] + rs * asn[i]);
    float hs = (1.0f - zs) * ns + zs * hold;

    const float hv = edges[gbase + d] ? hc : hs;
    h_out[(size_t)d * H + col]   = hv;
    hbf_out[(size_t)d * H + col] = (bf16_t)hv;
    if (outs) outs[(size_t)(gbase + d) * H + col] = (bf16_t)hv;
  }
}

// ---------------------------------------------------------------------------
// Small helper kernels
// ---------------------------------------------------------------------------
__global__ __launch_bounds__(256)
void f32_to_bf16_k(const float* __restrict__ src, bf16_t* __restrict__ dst, int n) {
  const int i = blockIdx.x * blockDim.x + threadIdx.x;
  if (i < n) dst[i] = (bf16_t)src[i];
}

__global__ __launch_bounds__(256)
void zero_bf16_k(bf16_t* __restrict__ dst, int n) {
  const int i = blockIdx.x * blockDim.x + threadIdx.x;
  if (i < n) dst[i] = (bf16_t)0.0f;
}

// dst[i] = i < n ? src[i] : 0   for i < npad
__global__ __launch_bounds__(256)
void copy_pad_f32_k(const float* __restrict__ src, float* __restrict__ dst,
                    int n, int npad) {
  const int i = blockIdx.x * blockDim.x + threadIdx.x;
  if (i < npad) dst[i] = (i < n) ? src[i] : 0.0f;
}

__global__ __launch_bounds__(256)
void gather_embed(const int* __restrict__ nodes, const float* __restrict__ emb,
                  bf16_t* __restrict__ Xbf) {
  using namespace cfg;
  const int g = blockIdx.x;                 // s*D + d
  const int node = nodes[g];
  const float* src = emb + (size_t)node * U;
  bf16_t* dst = Xbf + (size_t)g * U;
  for (int u = threadIdx.x; u < U; u += blockDim.x) dst[u] = (bf16_t)src[u];
}

__global__ __launch_bounds__(256)
void init_h(const float* __restrict__ src, float* __restrict__ h,
            bf16_t* __restrict__ hbf, int n) {
  const int i = blockIdx.x * blockDim.x + threadIdx.x;
  if (i < n) { const float v = src[i]; h[i] = v; hbf[i] = (bf16_t)v; }
}

// mean/logv/z: small fp32 GEMV-style kernel (D*L = 32768 threads, K = H)
__global__ __launch_bounds__(256)
void latent_mz(const float* __restrict__ hid,
               const float* __restrict__ Wm, const float* __restrict__ bm,
               const float* __restrict__ Wl, const float* __restrict__ bl,
               const float* __restrict__ eps,
               float* __restrict__ mean_out, float* __restrict__ logv_out,
               float* __restrict__ z_out, float* __restrict__ z_ws) {
  using namespace cfg;
  const int idx = blockIdx.x * blockDim.x + threadIdx.x;
  if (idx >= D * L) return;
  const int d = idx / L, l = idx % L;
  const float* hv = hid + (size_t)d * H;
  const float* wm = Wm + (size_t)l * H;
  const float* wl = Wl + (size_t)l * H;
  float sm = bm[l], sl = bl[l];
  for (int k = 0; k < H; ++k) { const float h = hv[k]; sm += h * wm[k]; sl += h * wl[k]; }
  const float z = eps[idx] * expf(0.5f * sl) + sm;
  mean_out[idx] = sm; logv_out[idx] = sl; z_out[idx] = z; z_ws[idx] = z;
}

// dec_init = z @ l2h_W^T + b  (D*H threads, K = L)
__global__ __launch_bounds__(256)
void dec_init_k(const float* __restrict__ z, const float* __restrict__ W,
                const float* __restrict__ b, float* __restrict__ h,
                bf16_t* __restrict__ hbf) {
  using namespace cfg;
  const int idx = blockIdx.x * blockDim.x + threadIdx.x;
  if (idx >= D * H) return;
  const int d = idx / H, j = idx % H;
  const float* zv = z + (size_t)d * L;
  const float* w  = W + (size_t)j * L;
  float s = b[j];
  for (int k = 0; k < L; ++k) s += zv[k] * w[k];
  h[idx] = s; hbf[idx] = (bf16_t)s;
}

// log_softmax: read padded logits row (stride VP, first V cols), write dense
// d_out row of V. One block per row.
__global__ __launch_bounds__(256)
void log_softmax_rows(const float* __restrict__ src, float* __restrict__ dst) {
  using namespace cfg;
  const float* p = src + (size_t)blockIdx.x * VP;
  float* q = dst + (size_t)blockIdx.x * V;
  __shared__ float red[256];
  float m = -__builtin_inff();
  for (int i = threadIdx.x; i < V; i += 256) m = fmaxf(m, p[i]);
  red[threadIdx.x] = m; __syncthreads();
  for (int s = 128; s > 0; s >>= 1) {
    if (threadIdx.x < s) red[threadIdx.x] = fmaxf(red[threadIdx.x], red[threadIdx.x + s]);
    __syncthreads();
  }
  m = red[0]; __syncthreads();
  float sum = 0.0f;
  for (int i = threadIdx.x; i < V; i += 256) sum += expf(p[i] - m);
  red[threadIdx.x] = sum; __syncthreads();
  for (int s = 128; s > 0; s >>= 1) {
    if (threadIdx.x < s) red[threadIdx.x] += red[threadIdx.x + s];
    __syncthreads();
  }
  const float lz = m + logf(red[0]);
  for (int i = threadIdx.x; i < V; i += 256) q[i] = p[i] - lz;
}

// ---------------------------------------------------------------------------
// Host launcher
// ---------------------------------------------------------------------------
extern "C" void kernel_launch(void* const* d_in, const int* in_sizes, int n_in,
                              void* d_out, int out_size, void* d_ws, size_t ws_size,
                              hipStream_t stream) {
  using namespace cfg;
  (void)in_sizes; (void)n_in; (void)out_size; (void)ws_size;

  const int*           nodes = (const int*)d_in[0];
  const unsigned char* edges = (const unsigned char*)d_in[1];   // jnp bool -> 1 byte
  const float*         emb   = (const float*)d_in[2];
  const float* Wih_f[4] = { (const float*)d_in[3],  (const float*)d_in[7],
                            (const float*)d_in[11], (const float*)d_in[15] };
  const float* Whh_f[4] = { (const float*)d_in[4],  (const float*)d_in[8],
                            (const float*)d_in[12], (const float*)d_in[16] };
  const float* bih[4]   = { (const float*)d_in[5],  (const float*)d_in[9],
                            (const float*)d_in[13], (const float*)d_in[17] };
  const float* bhh[4]   = { (const float*)d_in[6],  (const float*)d_in[10],
                            (const float*)d_in[14], (const float*)d_in[18] };
  const float* h2m_W = (const float*)d_in[19];
  const float* h2m_b = (const float*)d_in[20];
  const float* h2l_W = (const float*)d_in[21];
  const float* h2l_b = (const float*)d_in[22];
  const float* l2h_W = (const float*)d_in[23];
  const float* l2h_b = (const float*)d_in[24];
  const float* o2v_W = (const float*)d_in[25];
  const float* o2v_b = (const float*)d_in[26];
  const float* enc_init = (const float*)d_in[27];
  const float* eps      = (const float*)d_in[28];

  float* out_logp = (float*)d_out;
  float* out_mean = out_logp + (size_t)SD * V;
  float* out_logv = out_mean + (size_t)D * L;
  float* out_z    = out_logv + (size_t)D * L;

  char* wp = (char*)d_ws;
  auto take = [&wp](size_t bytes) -> char* {
    char* p = wp; wp += (bytes + 255) & ~(size_t)255; return p;
  };
  bf16_t* Xbf = (bf16_t*)take((size_t)SD * U * 2);
  bf16_t* WihB[4]; for (int i = 0; i < 4; ++i) WihB[i] = (bf16_t*)take((size_t)H3 * U * 2);
  bf16_t* WhhB[4]; for (int i = 0; i < 4; ++i) WhhB[i] = (bf16_t*)take((size_t)H3 * H * 2);
  bf16_t* o2vB   = (bf16_t*)take((size_t)VP * H * 2);   // padded rows zeroed
  float*  o2vbP  = (float*)take((size_t)VP * 4);        // padded bias
  float*  Gi[4]; for (int i = 0; i < 4; ++i) Gi[i] = (float*)take((size_t)SD * H3 * 4);
  bf16_t* outsB  = (bf16_t*)take((size_t)SD * H * 2);
  float*  logitsP = (float*)take((size_t)SD * VP * 4);  // padded logits staging
  float*  hbuf[2]; bf16_t* hbfbuf[2];
  for (int i = 0; i < 2; ++i) {
    hbuf[i]   = (float*)take((size_t)D * H * 4);
    hbfbuf[i] = (bf16_t*)take((size_t)D * H * 2);
  }
  float* z_ws = (float*)take((size_t)D * L * 4);

  // 1) convert weights to bf16 (then L2-resident for all GEMMs)
  for (int i = 0; i < 4; ++i) {
    int n = H3 * U;
    f32_to_bf16_k<<<(n + 255) / 256, 256, 0, stream>>>(Wih_f[i], WihB[i], n);
    n = H3 * H;
    f32_to_bf16_k<<<(n + 255) / 256, 256, 0, stream>>>(Whh_f[i], WhhB[i], n);
  }
  {
    int n = V * H;
    f32_to_bf16_k<<<(n + 255) / 256, 256, 0, stream>>>(o2v_W, o2vB, n);
    int npad = (VP - V) * H;   // zero the 16 padded vocab rows
    zero_bf16_k<<<(npad + 255) / 256, 256, 0, stream>>>(o2vB + (size_t)V * H, npad);
    copy_pad_f32_k<<<(VP + 255) / 256, 256, 0, stream>>>(o2v_b, o2vbP, V, VP);
  }

  // 2) gather X = emb[nodes] as bf16
  gather_embed<<<SD, 256, 0, stream>>>(nodes, emb, Xbf);

  // 3) precompute all input gates Gi = X @ Wih^T + bih (off critical path)
  for (int i = 0; i < 4; ++i)
    gemm_nt_bf16<<<dim3(H3 / 32, SD / 256), 128, 0, stream>>>(
        Xbf, WihB[i], bih[i], Gi[i], SD, H3, U, H3);

  // 4) encoder recurrence (ping-pong h buffers)
  init_h<<<(D * H + 255) / 256, 256, 0, stream>>>(enc_init, hbuf[0], hbfbuf[0], D * H);
  for (int s = 0; s < S; ++s) {
    const int pi = s & 1, po = pi ^ 1;
    rnn_step<<<H / 16, 256, 0, stream>>>(s, Gi[0], Gi[1], WhhB[0], WhhB[1],
                                         bhh[0], bhh[1], edges,
                                         hbuf[pi], hbfbuf[pi], hbuf[po], hbfbuf[po],
                                         (bf16_t*)nullptr);
  }
  // S even -> final hidden state is in buffer 0

  // 5) latent: mean/logv/z (to d_out + ws), then dec_init into h buffer 0
  latent_mz<<<(D * L + 255) / 256, 256, 0, stream>>>(
      hbuf[0], h2m_W, h2m_b, h2l_W, h2l_b, eps, out_mean, out_logv, out_z, z_ws);
  dec_init_k<<<(D * H + 255) / 256, 256, 0, stream>>>(z_ws, l2h_W, l2h_b,
                                                      hbuf[0], hbfbuf[0]);

  // 6) decoder recurrence, storing per-step outputs (bf16) for the logits GEMM
  for (int s = 0; s < S; ++s) {
    const int pi = s & 1, po = pi ^ 1;
    rnn_step<<<H / 16, 256, 0, stream>>>(s, Gi[2], Gi[3], WhhB[2], WhhB[3],
                                         bhh[2], bhh[3], edges,
                                         hbuf[pi], hbfbuf[pi], hbuf[po], hbfbuf[po],
                                         outsB);
  }

  // 7) logits = outs @ o2v_W^T + b into padded staging, then log_softmax
  gemm_nt_bf16<<<dim3(VP / 32, SD / 256), 128, 0, stream>>>(
      outsB, o2vB, o2vbP, logitsP, SD, VP, H, VP);
  log_softmax_rows<<<SD, 256, 0, stream>>>(logitsP, out_logp);
}
